// TransformerBlock_89885075571014
// MI455X (gfx1250) — matrix operand order repaired and checked by reading, not compiled, not run on hardware
//
#include <hip/hip_runtime.h>

// ---------------------------------------------------------------------------
// Transformer block for MI455X (gfx1250), wave32 + WMMA bf16 + async LDS copy
// with double-buffered (ping-pong) software pipeline.
//   B=4, S=2048, D_MODEL=1024, H=16, HEAD_DIM=64, D_INNER=4096
// ---------------------------------------------------------------------------

#define D_MODEL 1024
#define N_HEADS 16
#define HEAD_DIM 64
#define D_INNER 4096

#if __has_builtin(__builtin_amdgcn_global_load_async_to_lds_b128) && \
    __has_builtin(__builtin_amdgcn_s_wait_asynccnt)
#define USE_ASYNC_LDS 1
#else
#define USE_ASYNC_LDS 0
#endif

typedef __attribute__((ext_vector_type(16))) __bf16 v16bf;
typedef __attribute__((ext_vector_type(8)))  float  v8f;
typedef int v4i __attribute__((vector_size(16)));

union FragAB { v16bf v; unsigned int u[8]; };

#if USE_ASYNC_LDS
// one 16-byte async global->LDS copy (tracked by ASYNCcnt)
__device__ __forceinline__ void async_g2l_b128(const void* g, void* lds) {
  __builtin_amdgcn_global_load_async_to_lds_b128(
      (__attribute__((address_space(1))) v4i*)(unsigned long long)g,
      (__attribute__((address_space(3))) v4i*)(unsigned int)(unsigned long long)lds,
      0, 0);
}
#endif

// force global (AS1) loads instead of flat
__device__ __forceinline__ unsigned int ldg_u32(const void* p) {
  return *(const __attribute__((address_space(1))) unsigned int*)(unsigned long long)p;
}
__device__ __forceinline__ unsigned char ldg_u8(const void* p) {
  return *(const __attribute__((address_space(1))) unsigned char*)(unsigned long long)p;
}

// round-to-nearest-even f32 -> bf16 pair packed into one dword (v_perm_b32)
__device__ __forceinline__ unsigned int pack2_bf16(float a, float b) {
  unsigned int ua = __float_as_uint(a);
  unsigned int ub = __float_as_uint(b);
  ua += 0x7FFFu + ((ua >> 16) & 1u);
  ub += 0x7FFFu + ((ub >> 16) & 1u);
  return __builtin_amdgcn_perm(ub, ua, 0x07060302u);
}

__device__ __forceinline__ unsigned short bf16_1(float a) {
  unsigned int ua = __float_as_uint(a);
  ua += 0x7FFFu + ((ua >> 16) & 1u);
  return (unsigned short)(ua >> 16);
}

// 2x2 bf16 transpose: d0={A,B} (row k), d1={C,D} (row k+1) -> {A,C},{B,D}
__device__ __forceinline__ unsigned int trn_lo(unsigned int d0, unsigned int d1) {
  return __builtin_amdgcn_perm(d1, d0, 0x05040100u);
}
__device__ __forceinline__ unsigned int trn_hi(unsigned int d0, unsigned int d1) {
  return __builtin_amdgcn_perm(d1, d0, 0x07060302u);
}

// A-matrix (16x32 bf16) K-index for VGPR slot vv, lane half (ISA 7.12.2)
__device__ __forceinline__ int a_k0(int vv, int half) {
  return (vv < 4) ? (half * 8 + 2 * vv) : (16 + half * 8 + 2 * (vv - 4));
}

// ---------------------------------------------------------------------------
// f32 -> packed bf16, 8 elements per thread
// ---------------------------------------------------------------------------
__global__ __launch_bounds__(256)
void cvt_bf16_kernel(const float* __restrict__ src, unsigned short* __restrict__ dst) {
  const size_t base = ((size_t)blockIdx.x * 256 + threadIdx.x) * 8;
  const float4 f0 = *(const float4*)(src + base);
  const float4 f1 = *(const float4*)(src + base + 4);
  uint4 o;
  o.x = pack2_bf16(f0.x, f0.y);
  o.y = pack2_bf16(f0.z, f0.w);
  o.z = pack2_bf16(f1.x, f1.y);
  o.w = pack2_bf16(f1.z, f1.w);
  *(uint4*)(dst + base) = o;
}

// ---------------------------------------------------------------------------
// GEMM: C[M,N] = act(A[M,K] @ W[K,N]); A,W bf16; out f32 or bf16.
// mode: bit0 = relu, bit1 = bf16 output.
// Block 256 thr (8 waves); tile 256(M) x 64(N); K-step 32; double-buffered.
// ---------------------------------------------------------------------------
__global__ __launch_bounds__(256)
void gemm_bf16_wmma(const unsigned short* __restrict__ A,
                    const unsigned short* __restrict__ W,
                    void* __restrict__ Cout,
                    int M, int N, int K, int mode) {
  __shared__ __align__(16) unsigned short As[2][256 * 32]; // [m][k], ping-pong
  __shared__ __align__(16) unsigned short Bt[2][64 * 32];  // [n][k], ping-pong
  (void)M;

  const int tid  = threadIdx.x;
  const int w    = tid >> 5;
  const int l    = tid & 31;
  const int half = l >> 4;
  const int lan  = l & 15;
  const int mb   = blockIdx.y * 256;
  const int nb   = blockIdx.x * 64;

  // ---- staging descriptors ----
  const uint4* aSrc[4];
  uint4*       aDst[4];                 // buffer 0; buffer 1 = +1024 uint4
#pragma unroll
  for (int i = 0; i < 4; ++i) {
    const int id  = tid + 256 * i;
    const int row = id >> 2;            // 4 uint4 per 32-elem row
    const int q   = id & 3;
    aSrc[i] = (const uint4*)(A + (size_t)(mb + row) * K) + q;
    aDst[i] = (uint4*)&As[0][0] + id;
  }
  const unsigned int* bSrc0[2];
  const unsigned int* bSrc1[2];
  unsigned int*       bDst[2];          // buffer 0; buffer 1 = +1024 dwords
#pragma unroll
  for (int i = 0; i < 2; ++i) {
    const int unit = tid + 256 * i;
    const int kp = unit >> 5;           // k-pair 0..15
    const int np = unit & 31;           // n-pair 0..31
    bSrc0[i] = (const unsigned int*)(W + (size_t)(2 * kp) * N + nb) + np;
    bSrc1[i] = bSrc0[i] + (N >> 1);
    bDst[i]  = (unsigned int*)&Bt[0][0] + (2 * np) * 16 + kp;
  }
  const int bStep = N << 4;             // dwords per 32-row advance

  auto stage = [&](int sel) {
#if USE_ASYNC_LDS
#pragma unroll
    for (int i = 0; i < 4; ++i) async_g2l_b128(aSrc[i], aDst[i] + sel * 1024);
#else
    uint4 t[4];
#pragma unroll
    for (int i = 0; i < 4; ++i) t[i] = *aSrc[i];
#pragma unroll
    for (int i = 0; i < 4; ++i) *(aDst[i] + sel * 1024) = t[i];
#endif
    const unsigned int d00 = ldg_u32(bSrc0[0]), d10 = ldg_u32(bSrc1[0]);
    const unsigned int d01 = ldg_u32(bSrc0[1]), d11 = ldg_u32(bSrc1[1]);
    unsigned int* o0 = bDst[0] + sel * 1024;
    unsigned int* o1 = bDst[1] + sel * 1024;
    o0[0]  = trn_lo(d00, d10);
    o0[16] = trn_hi(d00, d10);
    o1[0]  = trn_lo(d01, d11);
    o1[16] = trn_hi(d01, d11);
#pragma unroll
    for (int i = 0; i < 4; ++i) aSrc[i] += 4;
#pragma unroll
    for (int i = 0; i < 2; ++i) { bSrc0[i] += bStep; bSrc1[i] += bStep; }
  };

  v8f acc[2][4] = {};

  stage(0);                              // prologue: tile 0 -> buffer 0
  int p = 0;
  for (int kb = 0; kb < K; kb += 32, p ^= 1) {
#if USE_ASYNC_LDS
    __builtin_amdgcn_s_wait_asynccnt(0);
#endif
    __syncthreads();                     // tile p visible; buffer p^1 free
    if (kb + 32 < K) {
      stage(p ^ 1);                      // overlaps with the WMMAs below
      __builtin_prefetch(aSrc[0], 0, 1);
      __builtin_prefetch(bSrc0[0], 0, 1);
    }

    // ---- fragments + 8 WMMAs from buffer p ----
    FragAB a0, a1, b;
    const int mrow0 = w * 32 + lan;
#pragma unroll
    for (int vv = 0; vv < 8; ++vv) {
      const int k0 = a_k0(vv, half);
      a0.u[vv] = *(const unsigned int*)&As[p][(mrow0)      * 32 + k0];
      a1.u[vv] = *(const unsigned int*)&As[p][(mrow0 + 16) * 32 + k0];
    }
#pragma unroll
    for (int nt = 0; nt < 4; ++nt) {
      const int n = nt * 16 + lan;
#pragma unroll
      for (int vv = 0; vv < 8; ++vv)
        b.u[vv] = *(const unsigned int*)&Bt[p][n * 32 + a_k0(vv, half)];
      acc[0][nt] = __builtin_amdgcn_wmma_f32_16x16x32_bf16(
          false, a0.v, false, b.v, (short)0, acc[0][nt], false, false);
      acc[1][nt] = __builtin_amdgcn_wmma_f32_16x16x32_bf16(
          false, a1.v, false, b.v, (short)0, acc[1][nt], false, false);
    }
  }

  // ---- epilogue: C/D layout M = r + 8*half, N = lan + 16*nt ----
  const size_t row0 = (size_t)mb + w * 32 + 8 * half;
  const size_t col0 = (size_t)nb + lan;
#pragma unroll
  for (int mi = 0; mi < 2; ++mi) {
#pragma unroll
    for (int nt = 0; nt < 4; ++nt) {
#pragma unroll
      for (int r = 0; r < 8; ++r) {
        float vv = acc[mi][nt][r];
        if (mode & 1) vv = fmaxf(vv, 0.0f);
        const size_t idx = (row0 + mi * 16 + r) * N + col0 + nt * 16;
        if (mode & 2) ((unsigned short*)Cout)[idx] = bf16_1(vv);
        else          ((float*)Cout)[idx] = vv;
      }
    }
  }
}

// ---------------------------------------------------------------------------
// Flash attention over bf16 q/k/v; bf16 ctx out. Double-buffered K/V chunks.
// One block = (b,h) x 64 query rows; 4 waves x 16 q-rows; keys in chunks of 32.
// ---------------------------------------------------------------------------
__global__ __launch_bounds__(128)
void attn_wmma(const unsigned short* __restrict__ Q,
               const unsigned short* __restrict__ Kb,
               const unsigned short* __restrict__ Vb,
               const unsigned char* __restrict__ mask,
               unsigned short* __restrict__ O, int S) {
  __shared__ __align__(16) unsigned short Ks[2][32 * 64];  // [key][d], ping-pong
  __shared__ __align__(16) unsigned short Vt[2][64 * 32];  // [d][key], ping-pong
  __shared__ __align__(16) unsigned short Pw[4][16 * 32];  // per-wave P scratch
  __shared__ float biasLds[2][32];

  const int tid  = threadIdx.x;
  const int w    = tid >> 5;
  const int l    = tid & 31;
  const int half = l >> 4;
  const int lan  = l & 15;
  const int bh   = blockIdx.y;
  const int b    = bh / N_HEADS;
  const int h    = bh % N_HEADS;
  const int q0   = blockIdx.x * 64 + w * 16;

  // ---- Q fragments (A layout), global dword gathers ----
  const unsigned short* qp = Q + (size_t)(b * S + q0 + lan) * D_MODEL + h * HEAD_DIM;
  FragAB aq[2];
#pragma unroll
  for (int f = 0; f < 2; ++f)
#pragma unroll
    for (int vv = 0; vv < 8; ++vv)
      aq[f].u[vv] = ldg_u32(qp + f * 32 + a_k0(vv, half));

  // ---- staging descriptors ----
  const uint4* kSrc[2];
  uint4*       kDst[2];                 // buffer 0; buffer 1 = +256 uint4
#pragma unroll
  for (int i = 0; i < 2; ++i) {
    const int id  = tid + 128 * i;
    const int row = id >> 3;            // 8 uint4 per 64-elem row
    const int q   = id & 7;
    kSrc[i] = (const uint4*)(Kb + (size_t)(b * S + row) * D_MODEL + h * HEAD_DIM) + q;
    kDst[i] = (uint4*)&Ks[0][0] + id;
  }
  const unsigned int* vSrc0[4];
  const unsigned int* vSrc1[4];
  unsigned int*       vDst[4];          // buffer 0; buffer 1 = +1024 dwords
#pragma unroll
  for (int i = 0; i < 4; ++i) {
    const int unit = tid + 128 * i;
    const int kp = unit >> 5;           // key-pair 0..15
    const int dp = unit & 31;           // d-pair 0..31
    vSrc0[i] = (const unsigned int*)(Vb + (size_t)(b * S + 2 * kp) * D_MODEL + h * HEAD_DIM) + dp;
    vSrc1[i] = vSrc0[i] + (D_MODEL >> 1);
    vDst[i]  = (unsigned int*)&Vt[0][0] + (2 * dp) * 16 + kp;
  }
  const unsigned char* mp = mask + (size_t)b * S + (tid & 31);
  const int kStep = 32 * D_MODEL / 8;   // uint4 per chunk advance
  const int vStep = 32 * D_MODEL / 2;   // dwords per chunk advance

  auto stage = [&](int sel) {
#if USE_ASYNC_LDS
#pragma unroll
    for (int i = 0; i < 2; ++i) async_g2l_b128(kSrc[i], kDst[i] + sel * 256);
#else
    uint4 tk[2];
#pragma unroll
    for (int i = 0; i < 2; ++i) tk[i] = *kSrc[i];
#pragma unroll
    for (int i = 0; i < 2; ++i) *(kDst[i] + sel * 256) = tk[i];
#endif
    unsigned int v0[4], v1[4];
#pragma unroll
    for (int i = 0; i < 4; ++i) { v0[i] = ldg_u32(vSrc0[i]); v1[i] = ldg_u32(vSrc1[i]); }
#pragma unroll
    for (int i = 0; i < 4; ++i) {
      unsigned int* o = vDst[i] + sel * 1024;
      o[0]  = trn_lo(v0[i], v1[i]);
      o[16] = trn_hi(v0[i], v1[i]);
    }
    if (tid < 32) biasLds[sel][tid] = ldg_u8(mp) ? 0.0f : -1e30f;
#pragma unroll
    for (int i = 0; i < 2; ++i) kSrc[i] += kStep;
#pragma unroll
    for (int i = 0; i < 4; ++i) { vSrc0[i] += vStep; vSrc1[i] += vStep; }
    mp += 32;
  };

  v8f ctx[4] = {};
  float mrow[8], lrow[8];
#pragma unroll
  for (int r = 0; r < 8; ++r) { mrow[r] = -1e30f; lrow[r] = 0.0f; }

  const float scale = 0.125f; // 64^-0.5

  stage(0);                              // prologue: chunk 0 -> buffer 0
  int p = 0;
  for (int kb = 0; kb < S; kb += 32, p ^= 1) {
#if USE_ASYNC_LDS
    __builtin_amdgcn_s_wait_asynccnt(0);
#endif
    __syncthreads();                     // chunk p visible; buffer p^1 free
    if (kb + 32 < S) stage(p ^ 1);       // overlaps with score/ctx WMMAs

    // ---- scores: two 16x16 tiles (keys 0-15 / 16-31), K-dim 64 = 2 WMMAs ----
    v8f sacc[2] = {};
#pragma unroll
    for (int kt = 0; kt < 2; ++kt) {
      const int key = kt * 16 + lan;
#pragma unroll
      for (int f = 0; f < 2; ++f) {
        FragAB bk;
#pragma unroll
        for (int vv = 0; vv < 8; ++vv)
          bk.u[vv] = *(const unsigned int*)&Ks[p][key * 64 + f * 32 + a_k0(vv, half)];
        sacc[kt] = __builtin_amdgcn_wmma_f32_16x16x32_bf16(
            false, aq[f].v, false, bk.v, (short)0, sacc[kt], false, false);
      }
    }

    // ---- online softmax; rows live in 16-lane halves ----
    const float bias0 = biasLds[p][lan];
    const float bias1 = biasLds[p][16 + lan];
#pragma unroll
    for (int r = 0; r < 8; ++r) {
      float s0 = sacc[0][r] * scale + bias0;
      float s1 = sacc[1][r] * scale + bias1;
      float mx = fmaxf(s0, s1);
#pragma unroll
      for (int off = 1; off < 16; off <<= 1)
        mx = fmaxf(mx, __shfl_xor(mx, off, 32));
      const float mnew = fmaxf(mrow[r], mx);
      const float corr = __expf(mrow[r] - mnew);
      const float p0 = __expf(s0 - mnew);
      const float p1 = __expf(s1 - mnew);
      float rs = p0 + p1;
#pragma unroll
      for (int off = 1; off < 16; off <<= 1)
        rs += __shfl_xor(rs, off, 32);
      lrow[r] = lrow[r] * corr + rs;
      mrow[r] = mnew;
#pragma unroll
      for (int nt = 0; nt < 4; ++nt) ctx[nt][r] *= corr;
      const int prow = r + 8 * half;
      Pw[w][prow * 32 + lan]      = bf16_1(p0);
      Pw[w][prow * 32 + 16 + lan] = bf16_1(p1);
    }
    __syncthreads();                     // publish P reshape

    // ---- ctx += P(16x32) x V(32x64) ----
    FragAB ap;
#pragma unroll
    for (int vv = 0; vv < 8; ++vv)
      ap.u[vv] = *(const unsigned int*)&Pw[w][lan * 32 + a_k0(vv, half)];
#pragma unroll
    for (int nt = 0; nt < 4; ++nt) {
      FragAB bv;
      const int d = nt * 16 + lan;
#pragma unroll
      for (int vv = 0; vv < 8; ++vv)
        bv.u[vv] = *(const unsigned int*)&Vt[p][d * 32 + a_k0(vv, half)];
      ctx[nt] = __builtin_amdgcn_wmma_f32_16x16x32_bf16(
          false, ap.v, false, bv.v, (short)0, ctx[nt], false, false);
    }
  }

  // ---- normalize, write bf16 context ----
#pragma unroll
  for (int r = 0; r < 8; ++r) {
    const float inv = 1.0f / lrow[r];
    const size_t row = (size_t)(b * S) + q0 + r + 8 * half;
#pragma unroll
    for (int nt = 0; nt < 4; ++nt)
      O[row * D_MODEL + h * HEAD_DIM + nt * 16 + lan] = bf16_1(ctx[nt][r] * inv);
  }
}

// ---------------------------------------------------------------------------
// out = (x+y) * rsqrt(mean((x+y)^2) + eps); optional packed-bf16 second output.
// ---------------------------------------------------------------------------
__global__ __launch_bounds__(256)
void addnorm_kernel(const float* __restrict__ X, const float* __restrict__ Y,
                    float* __restrict__ O, unsigned short* __restrict__ Ob) {
  __shared__ float wsum[8];
  const int row = blockIdx.x;
  const int tid = threadIdx.x;
  const size_t base = (size_t)row * D_MODEL + tid * 4;
  const float4 xv = *(const float4*)(X + base);
  const float4 yv = *(const float4*)(Y + base);
  float t0 = xv.x + yv.x, t1 = xv.y + yv.y, t2 = xv.z + yv.z, t3 = xv.w + yv.w;
  float ss = t0 * t0 + t1 * t1 + t2 * t2 + t3 * t3;
#pragma unroll
  for (int off = 1; off < 32; off <<= 1) ss += __shfl_xor(ss, off, 32);
  if ((tid & 31) == 0) wsum[tid >> 5] = ss;
  __syncthreads();
  float tot = 0.0f;
#pragma unroll
  for (int i = 0; i < 8; ++i) tot += wsum[i];
  const float rms = rsqrtf(tot * (1.0f / D_MODEL) + 1.1920929e-7f);
  float4 o;
  o.x = t0 * rms; o.y = t1 * rms; o.z = t2 * rms; o.w = t3 * rms;
  *(float4*)(O + base) = o;
  if (Ob) {
    uint2 p;
    p.x = pack2_bf16(o.x, o.y);
    p.y = pack2_bf16(o.z, o.w);
    *(uint2*)(Ob + base) = p;
  }
}

// ---------------------------------------------------------------------------
extern "C" void kernel_launch(void* const* d_in, const int* in_sizes, int n_in,
                              void* d_out, int out_size, void* d_ws, size_t ws_size,
                              hipStream_t stream) {
  (void)in_sizes; (void)n_in; (void)out_size; (void)ws_size;

  const float* x  = (const float*)d_in[0];
  const float* Wq = (const float*)d_in[1];
  const float* Wk = (const float*)d_in[2];
  const float* Wv = (const float*)d_in[3];
  const float* Wo = (const float*)d_in[4];
  const float* W1 = (const float*)d_in[5];
  const float* W2 = (const float*)d_in[6];
  const unsigned char* mask = (const unsigned char*)d_in[7];
  float* out = (float*)d_out;

  const int B = 4, S = 2048;
  const int Mtok = B * S;                        // 8192
  const size_t MB = 1u << 20;

  char* wsb = (char*)d_ws;
  unsigned short* wqb  = (unsigned short*)(wsb + 0 * MB);    //  2 MB
  unsigned short* wkb  = (unsigned short*)(wsb + 2 * MB);    //  2 MB
  unsigned short* wvb  = (unsigned short*)(wsb + 4 * MB);    //  2 MB
  unsigned short* wob  = (unsigned short*)(wsb + 6 * MB);    //  2 MB
  unsigned short* w1b  = (unsigned short*)(wsb + 8 * MB);    //  8 MB
  unsigned short* w2b  = (unsigned short*)(wsb + 16 * MB);   //  8 MB
  unsigned short* xb   = (unsigned short*)(wsb + 24 * MB);   // 16 MB
  unsigned short* qb   = (unsigned short*)(wsb + 40 * MB);   // 16 MB
  unsigned short* kbuf = (unsigned short*)(wsb + 56 * MB);   // 16 MB
  unsigned short* vbuf = (unsigned short*)(wsb + 72 * MB);   // 16 MB
  unsigned short* ctxb = (unsigned short*)(wsb + 88 * MB);   // 16 MB
  unsigned short* ff1b = (unsigned short*)(wsb + 40 * MB);   // 64 MB, aliases q/k/v/ctx (dead)
  float*          hbuf = (float*)(wsb + 104 * MB);           // 32 MB
  unsigned short* hb   = (unsigned short*)(wsb + 136 * MB);  // 16 MB
  float*          fscr = (float*)(wsb + 152 * MB);           // 32 MB (attn_out then ff2)

  const dim3 blk256(256);

  // ---- one-time f32 -> bf16 conversions ----
  cvt_bf16_kernel<<<dim3(512),  blk256, 0, stream>>>(Wq, wqb);
  cvt_bf16_kernel<<<dim3(512),  blk256, 0, stream>>>(Wk, wkb);
  cvt_bf16_kernel<<<dim3(512),  blk256, 0, stream>>>(Wv, wvb);
  cvt_bf16_kernel<<<dim3(512),  blk256, 0, stream>>>(Wo, wob);
  cvt_bf16_kernel<<<dim3(2048), blk256, 0, stream>>>(W1, w1b);
  cvt_bf16_kernel<<<dim3(2048), blk256, 0, stream>>>(W2, w2b);
  cvt_bf16_kernel<<<dim3(4096), blk256, 0, stream>>>(x,  xb);

  const dim3 gProj(D_MODEL / 64, Mtok / 256);    // (16, 32)
  const dim3 gFF1(D_INNER / 64, Mtok / 256);     // (64, 32)

  // ---- Q/K/V projections (bf16 out) ----
  gemm_bf16_wmma<<<gProj, blk256, 0, stream>>>(xb, wqb, qb,   Mtok, D_MODEL, D_MODEL, 2);
  gemm_bf16_wmma<<<gProj, blk256, 0, stream>>>(xb, wkb, kbuf, Mtok, D_MODEL, D_MODEL, 2);
  gemm_bf16_wmma<<<gProj, blk256, 0, stream>>>(xb, wvb, vbuf, Mtok, D_MODEL, D_MODEL, 2);

  // ---- flash attention (bf16 ctx out) ----
  const dim3 gAttn(S / 64, B * N_HEADS);         // (32, 64)
  attn_wmma<<<gAttn, dim3(128), 0, stream>>>(qb, kbuf, vbuf, mask, ctxb, S);

  // ---- output projection + residual RMSNorm (dual f32/bf16 h) ----
  gemm_bf16_wmma<<<gProj, blk256, 0, stream>>>(ctxb, wob, fscr, Mtok, D_MODEL, D_MODEL, 0);
  addnorm_kernel<<<dim3(Mtok), blk256, 0, stream>>>(x, fscr, hbuf, hb);

  // ---- FFN ----
  gemm_bf16_wmma<<<gFF1, blk256, 0, stream>>>(hb,   w1b, ff1b, Mtok, D_INNER, D_MODEL, 3);
  gemm_bf16_wmma<<<gProj, blk256, 0, stream>>>(ff1b, w2b, fscr, Mtok, D_MODEL, D_INNER, 0);

  // ---- final residual RMSNorm -> out ----
  addnorm_kernel<<<dim3(Mtok), blk256, 0, stream>>>(hbuf, fscr, out, nullptr);
}
